// MGGRUCell_80522046865918
// MI455X (gfx1250) — compile-verified, matrix operand order checked by reading, not run on hardware
//
#include <hip/hip_runtime.h>

// MGGRU cell forward for MI455X (gfx1250, wave32, WMMA).
// Heavy math (attention, corr, GCN propagation, gate linears) runs through a
// single templated WMMA GEMM (f16 in, f32 accumulate) with LDS tile staging.
// Each wave computes a 2x2 grid of 16x16 WMMA tiles (4 v_wmma per K-step per
// wave); the next K-tile is prefetched unconditionally (speculative
// global_prefetch_b8 is architecturally safe past the operand end).

typedef __attribute__((ext_vector_type(16))) _Float16 v16h;
typedef __attribute__((ext_vector_type(8)))  float    v8f;

#define DEV static __device__ __forceinline__

DEV float seluf(float x) {
  return 1.0507009873554805f * (x > 0.f ? x : 1.6732632423543772f * (__expf(x) - 1.f));
}
DEV float sigmf(float x) { return 1.f / (1.f + __expf(-x)); }
DEV float wave_sum(float v) {
#pragma unroll
  for (int m = 16; m >= 1; m >>= 1) v += __shfl_xor(v, m, 32);
  return v;
}
DEV float wave_max(float v) {
#pragma unroll
  for (int m = 16; m >= 1; m >>= 1) v = fmaxf(v, __shfl_xor(v, m, 32));
  return v;
}

// ---------------------------------------------------------------------------
// Generic WMMA GEMM:  C[M,N] (+)= scale * op(A)[M,K] * op(B)[K,N] + bias[N]
//   A_KM:  A stored [K,M] (compute with A transposed; used for A^T * h)
//   B_KN:  B stored [K,N] (normal NN); else B stored [N,K] (NT, e.g. Q*K^T)
//   OUT16: write _Float16, else float32 (with optional accumulate)
// Block: 256 threads = 8 waves as 4(M) x 2(N); each wave owns a 32x32 output
// (2x2 WMMA tiles), macro tile 128(M) x 64(N), K-step 32.
// All problem dims are multiples of the tiles (M%128==0, N%64==0, K%32==0).
// ---------------------------------------------------------------------------
template <bool A_KM, bool B_KN, bool OUT16, bool ACCUM>
__global__ __launch_bounds__(256) void wmma_gemm(
    const _Float16* __restrict__ A, int lda, long sA,
    const _Float16* __restrict__ Bm, int ldb, long sB,
    void* __restrict__ Cout, int ldc, long sC,
    const float* __restrict__ bias, float scale, int K) {
  __shared__ _Float16 As[128][40];  // [m][k] (+pad)
  __shared__ _Float16 Bs[64][40];   // [n][k] (+pad)
  const int bz = blockIdx.z;
  A  += (long)bz * sA;
  Bm += (long)bz * sB;
  const int m0 = blockIdx.y * 128;
  const int n0 = blockIdx.x * 64;
  const int t = threadIdx.x;
  const int lane = t & 31;
  const int wave = t >> 5;
  const int wr = wave >> 1;  // 0..3 -> M sub-tile (32 rows each)
  const int wc = wave & 1;   // 0..1 -> N sub-tile (32 cols each)
  v8f acc[2][2] = {};
  for (int k0 = 0; k0 < K; k0 += 32) {
    __syncthreads();
    if (!A_KM) {  // A row-major [M,K]: vectorizable row copies, 16 halves/thr
#pragma unroll
      for (int c = 0; c < 4; ++c) {
        const int idx = t + c * 256;
        const int row = idx >> 3, col = (idx & 7) * 4;
        const _Float16* src = A + (long)(m0 + row) * lda + (k0 + col);
#pragma unroll
        for (int i = 0; i < 4; ++i) As[row][col + i] = src[i];
        __builtin_prefetch((const void*)(src + 32), 0, 1);  // next K tile
      }
    } else {  // A stored [K,M]: transpose while staging
#pragma unroll
      for (int i = 0; i < 16; ++i) {
        const int e = t * 16 + i, mm = e >> 5, kk = e & 31;
        As[mm][kk] = A[(long)(k0 + kk) * lda + (m0 + mm)];
      }
    }
    if (B_KN) {  // B stored [K,N]: stage transposed -> Bs[n][k]
#pragma unroll
      for (int i = 0; i < 8; ++i) {
        const int e = t * 8 + i, kk = e >> 6, nn = e & 63;
        Bs[nn][kk] = Bm[(long)(k0 + kk) * ldb + (n0 + nn)];
      }
      __builtin_prefetch((const void*)(Bm + (long)(k0 + 32 + (t >> 6)) * ldb + n0 + (t & 63)), 0, 1);
    } else {  // B stored [N,K]: straight row copies
      const int nn = t >> 2, col = (t & 3) * 8;
      const _Float16* src = Bm + (long)(n0 + nn) * ldb + (k0 + col);
#pragma unroll
      for (int i = 0; i < 8; ++i) Bs[nn][col + i] = src[i];
      __builtin_prefetch((const void*)(src + 32), 0, 1);  // next K tile
    }
    __syncthreads();
    // Fragments per the 16-bit 16x32 A layout (lanes 0-15: K 0..7 / 16..23
    // halves; lanes 16-31: K 8..15 / 24..31). B symmetric over N.
    const int fm = lane & 15, g = lane >> 4;
    v16h af[2], bf[2];
#pragma unroll
    for (int mt = 0; mt < 2; ++mt) {
#pragma unroll
      for (int v = 0; v < 8; ++v) {
        const int kk = ((v & 4) << 2) + g * 8 + ((v & 3) << 1);
        af[mt][2 * v]     = As[wr * 32 + mt * 16 + fm][kk];
        af[mt][2 * v + 1] = As[wr * 32 + mt * 16 + fm][kk + 1];
        bf[mt][2 * v]     = Bs[wc * 32 + mt * 16 + fm][kk];
        bf[mt][2 * v + 1] = Bs[wc * 32 + mt * 16 + fm][kk + 1];
      }
    }
#pragma unroll
    for (int mt = 0; mt < 2; ++mt)
#pragma unroll
      for (int nt = 0; nt < 2; ++nt)
        acc[mt][nt] = __builtin_amdgcn_wmma_f32_16x16x32_f16(
            false, af[mt], false, bf[nt], (short)0, acc[mt][nt], false, false);
  }
  // C layout: VGPR r -> M = r (+8 for lanes 16-31), N = lane&15
#pragma unroll
  for (int mt = 0; mt < 2; ++mt) {
#pragma unroll
    for (int nt = 0; nt < 2; ++nt) {
      const int n = n0 + wc * 32 + nt * 16 + (lane & 15);
      const int mb = m0 + wr * 32 + mt * 16 + (lane >> 4) * 8;
      const float bv = bias ? bias[n] : 0.f;
#pragma unroll
      for (int r = 0; r < 8; ++r) {
        const float val = acc[mt][nt][r] * scale + bv;
        const long off = (long)bz * sC + (long)(mb + r) * ldc + n;
        if (OUT16) {
          ((_Float16*)Cout)[off] = (_Float16)val;
        } else {
          float* o = (float*)Cout;
          o[off] = ACCUM ? (o[off] + val) : val;
        }
      }
    }
  }
}

// ---------------------------------------------------------------------------
// Elementwise / row kernels
// ---------------------------------------------------------------------------
__global__ void k_cast(const float* __restrict__ s, _Float16* __restrict__ d, int n) {
  const int i = blockIdx.x * 256 + threadIdx.x;
  if (i < n) d[i] = (_Float16)s[i];
}

// In-place row softmax over 1024 f16 columns, one wave per row.
__global__ void k_softmax(_Float16* __restrict__ p) {
  const long row = (long)blockIdx.x * 8 + (threadIdx.x >> 5);
  const int lane = threadIdx.x & 31;
  _Float16* r = p + row * 1024;
  float v[32];
#pragma unroll
  for (int j = 0; j < 32; ++j) v[j] = (float)r[lane + j * 32];
  float m = -1e30f;
#pragma unroll
  for (int j = 0; j < 32; ++j) m = fmaxf(m, v[j]);
  m = wave_max(m);
  float s = 0.f;
#pragma unroll
  for (int j = 0; j < 32; ++j) { v[j] = __expf(v[j] - m); s += v[j]; }
  const float inv = 1.f / wave_sum(s);
#pragma unroll
  for (int j = 0; j < 32; ++j) r[lane + j * 32] = (_Float16)(v[j] * inv);
}

// adj = relu(tanh(corr)); norm = adj / (rowsum + 1e-6). f32 + f16 outputs.
__global__ void k_adjnorm(const float* __restrict__ corr, float* __restrict__ nf,
                          _Float16* __restrict__ n16) {
  const long row = (long)blockIdx.x * 8 + (threadIdx.x >> 5);
  const int lane = threadIdx.x & 31;
  const float* cr = corr + row * 1024;
  float v[32];
  float s = 0.f;
#pragma unroll
  for (int j = 0; j < 32; ++j) {
    float a = tanhf(cr[lane + j * 32]);
    a = a > 0.f ? a : 0.f;
    v[j] = a; s += a;
  }
  const float inv = 1.f / (wave_sum(s) + 1e-6f);
#pragma unroll
  for (int j = 0; j < 32; ++j) {
    const float nv = v[j] * inv;
    nf[row * 1024 + lane + j * 32] = nv;
    n16[row * 1024 + lane + j * 32] = (_Float16)nv;
  }
}

// x1 = LN(selu(x + selu(attn))); also writes combined=[x1,H] (f16) into concat.
__global__ void k_post_attn1(const float* __restrict__ x, const float* __restrict__ at,
                             const float* __restrict__ H, const float* __restrict__ g,
                             const float* __restrict__ b, float* __restrict__ x1,
                             _Float16* __restrict__ concat) {
  const long row = (long)blockIdx.x * 8 + (threadIdx.x >> 5);
  const int lane = threadIdx.x & 31;
  const int c0 = lane, c1 = lane + 32;
  const long base = row * 64;
  const float t0 = seluf(x[base + c0] + seluf(at[base + c0]));
  const float t1 = seluf(x[base + c1] + seluf(at[base + c1]));
  const float mean = wave_sum(t0 + t1) * (1.f / 64.f);
  const float d0 = t0 - mean, d1 = t1 - mean;
  const float var = wave_sum(d0 * d0 + d1 * d1) * (1.f / 64.f);
  const float rs = rsqrtf(var + 1e-5f);
  const float y0 = d0 * rs * g[c0] + b[c0];
  const float y1 = d1 * rs * g[c1] + b[c1];
  x1[base + c0] = y0; x1[base + c1] = y1;
  const long cb = row * 896;
  concat[cb + c0] = (_Float16)y0;
  concat[cb + c1] = (_Float16)y1;
  concat[cb + 64 + c0] = (_Float16)H[base + c0];
  concat[cb + 64 + c1] = (_Float16)H[base + c1];
}

// Bottleneck embedding: e = tanh(selu(z*W1+b1)*W2+b2), z = combined row (128).
__global__ void k_adap(const _Float16* __restrict__ concat,
                       const float* __restrict__ W1, const float* __restrict__ b1,
                       const float* __restrict__ W2, const float* __restrict__ b2,
                       _Float16* __restrict__ e16) {
  __shared__ float z[128];
  __shared__ float s4[4];
  const long node = blockIdx.x;
  const int t = threadIdx.x;  // 64 threads
  const _Float16* row = concat + node * 896;
  z[t] = (float)row[t];
  z[t + 64] = (float)row[t + 64];
  __syncthreads();
  if (t < 4) {
    float s = b1[t];
    for (int k = 0; k < 128; ++k) s += z[k] * W1[k * 4 + t];
    s4[t] = seluf(s);
  }
  __syncthreads();
  float v = b2[t];
#pragma unroll
  for (int j = 0; j < 4; ++j) v += s4[j] * W2[j * 64 + t];
  e16[node * 64 + t] = (_Float16)tanhf(v);
}

// u=sigmoid(upd); r=sigmoid(rst); Hs=selu(H); concat[:,64:128] = r*Hs (f16).
__global__ void k_gates(float* __restrict__ upd, const float* __restrict__ rst,
                        const float* __restrict__ H, _Float16* __restrict__ concat,
                        float* __restrict__ Hs, int n) {
  const int i = blockIdx.x * 256 + threadIdx.x;
  if (i >= n) return;
  const float u = sigmf(upd[i]);
  const float r = sigmf(rst[i]);
  const float hs = seluf(H[i]);
  upd[i] = u;
  Hs[i] = hs;
  const long node = (long)(i >> 6);
  const int c = i & 63;
  concat[node * 896 + 64 + c] = (_Float16)(r * hs);
}

// nH = u*Hs + (1-u)*tanh(cand); write f32 + f16.
__global__ void k_candout(const float* __restrict__ cand, const float* __restrict__ u,
                          const float* __restrict__ Hs, float* __restrict__ nH,
                          _Float16* __restrict__ nH16, int n) {
  const int i = blockIdx.x * 256 + threadIdx.x;
  if (i >= n) return;
  const float c = tanhf(cand[i]);
  const float v = u[i] * Hs[i] + (1.f - u[i]) * c;
  nH[i] = v;
  nH16[i] = (_Float16)v;
}

// y = LN(selu(x1 + selu(attnB))); next = 0.3*Hs + 0.7*y; out = selu(y + x1).
__global__ void k_post_attn2(const float* __restrict__ x1, const float* __restrict__ at,
                             const float* __restrict__ Hs, const float* __restrict__ g,
                             const float* __restrict__ b, float* __restrict__ outO,
                             float* __restrict__ outN) {
  const long row = (long)blockIdx.x * 8 + (threadIdx.x >> 5);
  const int lane = threadIdx.x & 31;
  const int c0 = lane, c1 = lane + 32;
  const long base = row * 64;
  const float x0 = x1[base + c0], xx1 = x1[base + c1];
  const float t0 = seluf(x0 + seluf(at[base + c0]));
  const float t1 = seluf(xx1 + seluf(at[base + c1]));
  const float mean = wave_sum(t0 + t1) * (1.f / 64.f);
  const float d0 = t0 - mean, d1 = t1 - mean;
  const float var = wave_sum(d0 * d0 + d1 * d1) * (1.f / 64.f);
  const float rs = rsqrtf(var + 1e-5f);
  const float y0 = d0 * rs * g[c0] + b[c0];
  const float y1 = d1 * rs * g[c1] + b[c1];
  outN[base + c0] = 0.3f * Hs[base + c0] + 0.7f * y0;
  outN[base + c1] = 0.3f * Hs[base + c1] + 0.7f * y1;
  outO[base + c0] = seluf(y0 + x0);
  outO[base + c1] = seluf(y1 + xx1);
}

// ---------------------------------------------------------------------------
// Host-side orchestration
// ---------------------------------------------------------------------------
template <bool AKM, bool BKN, bool O16, bool ACC>
static void gemm(const _Float16* A, int lda, long sA, const _Float16* Bm, int ldb,
                 long sB, void* C, int ldc, long sC, const float* bias, float scale,
                 int M, int N, int K, int batch, hipStream_t st) {
  dim3 g(N / 64, M / 128, batch), blk(256, 1, 1);
  wmma_gemm<AKM, BKN, O16, ACC><<<g, blk, 0, st>>>(A, lda, sA, Bm, ldb, sB, C, ldc,
                                                   sC, bias, scale, K);
}

extern "C" void kernel_launch(void* const* d_in, const int* in_sizes, int n_in,
                              void* d_out, int out_size, void* d_ws, size_t ws_size,
                              hipStream_t stream) {
  (void)in_sizes; (void)n_in; (void)out_size; (void)ws_size;
  constexpr long BN = 16384, BNC = 1048576L, NN = 1048576L, BNN = 16777216L;

  // ---- inputs (setup_inputs order, params dict flattened in insertion order)
  const float* x    = (const float*)d_in[0];
  const float* H    = (const float*)d_in[2];
  const float* adj1 = (const float*)d_in[3];
  const float* adj2 = (const float*)d_in[4];
  const float* Wat[6]; const float* bat[6];
  for (int i = 0; i < 6; ++i) { Wat[i] = (const float*)d_in[5 + 2 * i]; bat[i] = (const float*)d_in[6 + 2 * i]; }
  const float* ln_g = (const float*)d_in[17];
  const float* ln_b = (const float*)d_in[18];
  const float* Wf1 = (const float*)d_in[19]; const float* bf1 = (const float*)d_in[20];
  const float* Wf2 = (const float*)d_in[21]; const float* bf2 = (const float*)d_in[22];
  const float* Wg[6]; const float* bg[6];  // u1,u2,r1,r2,c1,c2
  for (int i = 0; i < 6; ++i) { Wg[i] = (const float*)d_in[23 + 2 * i]; bg[i] = (const float*)d_in[24 + 2 * i]; }

  // ---- outputs: (output, next_hidden, norm_adj, corr)
  float* out      = (float*)d_out;
  float* out_out  = out;
  float* out_nh   = out + BNC;
  float* out_norm = out + 2 * BNC;
  float* out_corr = out + 2 * BNC + BNN;

  // ---- workspace arena
  char* w = (char*)d_ws;
  size_t o = 0;
  auto take = [&](size_t bytes) -> void* {
    void* p = w + o; o = (o + bytes + 255) & ~(size_t)255; return p;
  };
  _Float16* x16  = (_Float16*)take(BNC * 2);
  _Float16* a1h  = (_Float16*)take(NN * 2);
  _Float16* a2h  = (_Float16*)take(NN * 2);
  _Float16* wat16[6]; for (int i = 0; i < 6; ++i) wat16[i] = (_Float16*)take(4096 * 2);
  _Float16* wg16[6];  for (int i = 0; i < 6; ++i) wg16[i]  = (_Float16*)take(57344 * 2);
  _Float16* Q16 = (_Float16*)take(BNC * 2);
  _Float16* K16 = (_Float16*)take(BNC * 2);
  _Float16* V16 = (_Float16*)take(BNC * 2);
  _Float16* big16 = (_Float16*)take(BNN * 2);          // scoresA -> norm_adj16 -> scoresB
  _Float16* concat16 = (_Float16*)take(BN * 896 * 2);  // [16384, 896] hop concat
  float* attnb = (float*)take(BNC * 4);
  float* x1    = (float*)take(BNC * 4);
  _Float16* e16 = (_Float16*)take(BNC * 2);
  float* upd = (float*)take(BNC * 4);
  float* rst = (float*)take(BNC * 4);
  float* cnd = (float*)take(BNC * 4);
  float* Hs  = (float*)take(BNC * 4);
  float* nH  = (float*)take(BNC * 4);
  _Float16* nH16 = (_Float16*)take(BNC * 2);

  auto cast = [&](const float* s, _Float16* d, long n) {
    k_cast<<<dim3((unsigned)((n + 255) / 256)), dim3(256), 0, stream>>>(s, d, (int)n);
  };

  // ---- precision converts
  cast(x, x16, BNC);
  cast(adj1, a1h, NN);
  cast(adj2, a2h, NN);
  for (int i = 0; i < 6; ++i) cast(Wat[i], wat16[i], 4096);
  for (int i = 0; i < 6; ++i) cast(Wg[i], wg16[i], 57344);

  // ---- self-attention block (shared by A and B)
  auto attention = [&](const _Float16* in16, int wbase, float* attn_out) {
    gemm<false, true, true, false>(in16, 64, 0, wat16[wbase + 0], 64, 0, Q16, 64, 0, bat[wbase + 0], 1.f, (int)BN, 64, 64, 1, stream);
    gemm<false, true, true, false>(in16, 64, 0, wat16[wbase + 1], 64, 0, K16, 64, 0, bat[wbase + 1], 1.f, (int)BN, 64, 64, 1, stream);
    gemm<false, true, true, false>(in16, 64, 0, wat16[wbase + 2], 64, 0, V16, 64, 0, bat[wbase + 2], 1.f, (int)BN, 64, 64, 1, stream);
    // scores = Q*K^T / sqrt(C)   (per batch)
    gemm<false, false, true, false>(Q16, 64, 65536L, K16, 64, 65536L, big16, 1024, 1048576L, nullptr, 0.125f, 1024, 1024, 64, 16, stream);
    k_softmax<<<2048, 256, 0, stream>>>(big16);
    gemm<false, true, false, false>(big16, 1024, 1048576L, V16, 64, 65536L, attn_out, 64, 65536L, nullptr, 1.f, 1024, 64, 1024, 16, stream);
  };

  attention(x16, 0, attnb);
  k_post_attn1<<<2048, 256, 0, stream>>>(x, attnb, H, ln_g, ln_b, x1, concat16);

  // ---- adaptive adjacency
  k_adap<<<16384, 64, 0, stream>>>(concat16, Wf1, bf1, Wf2, bf2, e16);
  gemm<false, false, false, false>(e16, 64, 65536L, e16, 64, 65536L, out_corr, 1024, 1048576L, nullptr, 1.f, 1024, 1024, 64, 16, stream);
  k_adjnorm<<<2048, 256, 0, stream>>>(out_corr, out_norm, big16);

  // ---- GCN hop builders: slices 1..6 = {A1 h, A1 h2, A2 h, A2 h2, Ad h, Ad h2}
  const _Float16* sup[3] = {a1h, a2h, big16};
  const long supS[3] = {0, 0, NN};
  const long cs = 896L * 1024;  // per-batch stride of concat16
  auto hops_fwd = [&]() {
    for (int s = 0; s < 3; ++s) {
      const int o1 = 2 * s + 1, o2 = 2 * s + 2;
      gemm<false, true, true, false>(sup[s], 1024, supS[s], concat16, 896, cs, concat16 + o1 * 128, 896, cs, nullptr, 1.f, 1024, 128, 1024, 16, stream);
      gemm<false, true, true, false>(sup[s], 1024, supS[s], concat16 + o1 * 128, 896, cs, concat16 + o2 * 128, 896, cs, nullptr, 1.f, 1024, 128, 1024, 16, stream);
    }
  };
  auto hops_t = [&]() {
    for (int s = 0; s < 3; ++s) {
      const int o1 = 2 * s + 1, o2 = 2 * s + 2;
      gemm<true, true, true, false>(sup[s], 1024, supS[s], concat16, 896, cs, concat16 + o1 * 128, 896, cs, nullptr, 1.f, 1024, 128, 1024, 16, stream);
      gemm<true, true, true, false>(sup[s], 1024, supS[s], concat16 + o1 * 128, 896, cs, concat16 + o2 * 128, 896, cs, nullptr, 1.f, 1024, 128, 1024, 16, stream);
    }
  };

  // ---- update / reset gates
  hops_fwd();
  gemm<false, true, false, false>(concat16, 896, 0, wg16[0], 64, 0, upd, 64, 0, bg[0], 1.f, (int)BN, 64, 896, 1, stream);  // u1
  gemm<false, true, false, false>(concat16, 896, 0, wg16[2], 64, 0, rst, 64, 0, bg[2], 1.f, (int)BN, 64, 896, 1, stream);  // r1
  hops_t();
  gemm<false, true, false, true>(concat16, 896, 0, wg16[1], 64, 0, upd, 64, 0, bg[1], 1.f, (int)BN, 64, 896, 1, stream);   // u2 (+=)
  gemm<false, true, false, true>(concat16, 896, 0, wg16[3], 64, 0, rst, 64, 0, bg[3], 1.f, (int)BN, 64, 896, 1, stream);   // r2 (+=)
  k_gates<<<4096, 256, 0, stream>>>(upd, rst, H, concat16, Hs, (int)BNC);

  // ---- candidate
  hops_fwd();
  gemm<false, true, false, false>(concat16, 896, 0, wg16[4], 64, 0, cnd, 64, 0, bg[4], 1.f, (int)BN, 64, 896, 1, stream);  // c1
  hops_t();
  gemm<false, true, false, true>(concat16, 896, 0, wg16[5], 64, 0, cnd, 64, 0, bg[5], 1.f, (int)BN, 64, 896, 1, stream);   // c2 (+=)
  k_candout<<<4096, 256, 0, stream>>>(cnd, upd, Hs, nH, nH16, (int)BNC);

  // ---- second attention block + outputs
  attention(nH16, 3, attnb);
  k_post_attn2<<<2048, 256, 0, stream>>>(x1, attnb, Hs, ln_g, ln_b, out_out, out_nh);
}
